// RegionProposalNetwork_84482006713118
// MI455X (gfx1250) — compile-verified, matrix operand order checked by reading, not run on hardware
//
#include <hip/hip_runtime.h>
#include <hip/hip_bf16.h>

typedef __bf16 bhalf;
typedef __attribute__((ext_vector_type(8)))  __bf16 v8bf;
typedef __attribute__((ext_vector_type(16))) __bf16 v16bf;
typedef __attribute__((ext_vector_type(8)))  float  v8f;
typedef int v4i_t __attribute__((vector_size(16)));   // element type the async builtin wants

// Problem constants
#define BATCH   8
#define CIN     2048
#define HH      32
#define WW      32
#define HID     512
#define HP      34          // padded H
#define WP      34          // padded W
#define LDA     40          // LDS row stride in bf16 elems (80 B: 16B-aligned, bank-spread)
#define NKSTEPS 576         // 9 taps * 64 channel-blocks of 32

// CDNA5 async global->LDS path (ASYNCcnt), if this toolchain exposes the builtins.
#if __has_builtin(__builtin_amdgcn_global_load_async_to_lds_b128) && \
    __has_builtin(__builtin_amdgcn_s_wait_asynccnt)
#define USE_ASYNC_LDS 1
#else
#define USE_ASYNC_LDS 0
#endif

#if USE_ASYNC_LDS
typedef __attribute__((address_space(1))) v4i_t* gv4p;   // global, 64-bit
typedef __attribute__((address_space(3))) v4i_t* lv4p;   // LDS, 32-bit
__device__ __forceinline__ void async_b128(const bhalf* g, bhalf* l) {
    // Flat->AS casts via integers: global addresses are identical in AS1;
    // low 32 bits of a flat LDS address are the LDS byte offset (aperture scheme).
    __builtin_amdgcn_global_load_async_to_lds_b128(
        (gv4p)(unsigned long long)(uintptr_t)g,
        (lv4p)(unsigned int)(uintptr_t)l,
        0, 0);
}
#endif

// ---------------------------------------------------------------------------
// Kernel 0: zero the padded input buffer (borders must be zero).
__global__ __launch_bounds__(256) void k_zero(uint4* __restrict__ p, int n16) {
    int i = blockIdx.x * 256 + threadIdx.x;
    if (i < n16) {
        uint4 z; z.x = 0; z.y = 0; z.z = 0; z.w = 0;
        p[i] = z;
    }
}

// ---------------------------------------------------------------------------
// Kernel 1: NCHW fp32 -> padded NHWC bf16.
// One wave per (b, y, 8-channel block); lanes = x (coalesced 128B reads per channel).
__global__ __launch_bounds__(256) void k_pack_input(const float* __restrict__ fm,
                                                    bhalf* __restrict__ Apad) {
    int wid  = blockIdx.x * 8 + (threadIdx.x >> 5);   // 65536 waves total
    int lane = threadIdx.x & 31;                       // x coordinate
    int cblk = wid & 255;                              // 256 blocks of 8 channels
    int y    = (wid >> 8) & 31;
    int b    = wid >> 13;
    int c0   = cblk * 8;
    v8bf o;
#pragma unroll
    for (int j = 0; j < 8; ++j) {
        int c = c0 + j;
        float v = fm[((b * CIN + c) * HH + y) * WW + lane];
        o[j] = (bhalf)v;
    }
    bhalf* dst = Apad + (((b * HP) + (y + 1)) * WP + (lane + 1)) * CIN + c0;
    *(v8bf*)dst = o;   // 16B aligned: pixel*4096 + cblk*16
}

// ---------------------------------------------------------------------------
// Kernel 2: W_hid (512,2048,3,3) fp32 -> Wpk[tap][cblk(64)][n(512)][ci(32)] bf16.
__global__ __launch_bounds__(256) void k_pack_weight(const float* __restrict__ Wh,
                                                     bhalf* __restrict__ Wpk) {
    int o  = blockIdx.x * 256 + threadIdx.x;   // exactly 9437184 threads
    int ci = o & 31;
    int n  = (o >> 5) & 511;
    int cb = (o >> 14) & 63;
    int t  = o >> 20;
    int c  = cb * 32 + ci;
    Wpk[o] = (bhalf)Wh[(n * CIN + c) * 9 + t];
}

// ---------------------------------------------------------------------------
// Kernel 3: implicit-GEMM 3x3 conv + bias + ReLU via bf16 WMMA, fp32 accum.
// Block tile 128(M) x 128(N); 8 waves as 4(M) x 2(N); wave tile 32 x 64.
__global__ __launch_bounds__(256) void k_conv1_wmma(const bhalf* __restrict__ Apad,
                                                    const bhalf* __restrict__ Wpk,
                                                    const float* __restrict__ bias,
                                                    bhalf* __restrict__ Xh) {
    __shared__ alignas(16) bhalf Al[2][128 * LDA];
    __shared__ alignas(16) bhalf Bl[2][128 * LDA];

    const int tid  = threadIdx.x;
    const int m0   = blockIdx.x * 128;
    const int n0   = blockIdx.y * 128;
    const int lane = tid & 31;
    const int wv   = tid >> 5;
    const int wm   = (wv & 3) * 32;
    const int wn   = (wv >> 2) * 64;
    const int bimg = m0 >> 10;              // 1024 pixels per image; 128 | 1024

    // Each thread moves 2x16B chunks of A and of B per k-step.
    const int r0  = tid >> 2;               // rows 0..63
    const int r1  = r0 + 64;                // rows 64..127
    const int cb0 = (tid & 3) * 8;          // element column (0,8,16,24)

    v8f acc[2][4];
    const v8f zf = {0.f, 0.f, 0.f, 0.f, 0.f, 0.f, 0.f, 0.f};
#pragma unroll
    for (int i = 0; i < 2; ++i)
#pragma unroll
        for (int j = 0; j < 4; ++j) acc[i][j] = zf;

    // Global source addresses for one k-step (tap, channel block).
    auto gsrc = [&](int ks, int r, const bhalf*& ga, const bhalf*& gb) {
        const int tap = ks >> 6;
        const int kc  = ks & 63;
        const int dy  = tap / 3, dx = tap % 3;
        const int c0  = kc * 32;
        int p  = m0 + r;
        int yy = ((p & 1023) >> 5) + dy;
        int xx = (p & 31) + dx;
        ga = Apad + (((bimg * HP + yy) * WP + xx) * CIN + c0 + cb0);
        gb = Wpk + ((tap * 64 + kc) * HID + n0 + r) * 32 + cb0;
    };

#if USE_ASYNC_LDS
    auto issue = [&](int ks, int buf) {
        const bhalf *ga0, *gb0, *ga1, *gb1;
        gsrc(ks, r0, ga0, gb0);
        gsrc(ks, r1, ga1, gb1);
        async_b128(ga0, Al[buf] + r0 * LDA + cb0);
        async_b128(ga1, Al[buf] + r1 * LDA + cb0);
        async_b128(gb0, Bl[buf] + r0 * LDA + cb0);
        async_b128(gb1, Bl[buf] + r1 * LDA + cb0);
    };
    issue(0, 0);
#else
    uint4 ra0, ra1, rb0, rb1;
    auto fetch = [&](int ks) {
        const bhalf *ga0, *gb0, *ga1, *gb1;
        gsrc(ks, r0, ga0, gb0);
        gsrc(ks, r1, ga1, gb1);
        ra0 = *(const uint4*)ga0;
        ra1 = *(const uint4*)ga1;
        rb0 = *(const uint4*)gb0;
        rb1 = *(const uint4*)gb1;
    };
    auto stash = [&](int buf) {
        *(uint4*)(Al[buf] + r0 * LDA + cb0) = ra0;
        *(uint4*)(Al[buf] + r1 * LDA + cb0) = ra1;
        *(uint4*)(Bl[buf] + r0 * LDA + cb0) = rb0;
        *(uint4*)(Bl[buf] + r1 * LDA + cb0) = rb1;
    };
    fetch(0);
    stash(0);
#endif

    // 16-bit A-fragment layout (ISA 7.12.2): lane<16 -> K{0..7,16..23}, lane>=16 -> K{8..15,24..31}
    const int lrow = lane & 15;
    const int kh   = (lane >> 4) * 8;

    for (int ks = 0; ks < NKSTEPS; ++ks) {
        const int cur = ks & 1;
#if USE_ASYNC_LDS
        if (ks + 1 < NKSTEPS) {
            issue(ks + 1, (ks + 1) & 1);              // fills the other buffer
            __builtin_amdgcn_s_wait_asynccnt(4);       // tile ks resident (in-order)
        } else {
            __builtin_amdgcn_s_wait_asynccnt(0);
        }
        __syncthreads();                               // publish all waves' portions
#else
        if (ks + 1 < NKSTEPS) fetch(ks + 1);
        __syncthreads();                               // buf[cur] fully populated
#endif
        const bhalf* Ab = Al[cur];
        const bhalf* Bb = Bl[cur];

        v16bf afr[2], bfr[4];
#pragma unroll
        for (int i = 0; i < 2; ++i) {
            const bhalf* p = Ab + (wm + i * 16 + lrow) * LDA + kh;
            v8bf lo = *(const v8bf*)p;
            v8bf hi = *(const v8bf*)(p + 16);
            afr[i] = __builtin_shufflevector(lo, hi, 0, 1, 2, 3, 4, 5, 6, 7,
                                                     8, 9, 10, 11, 12, 13, 14, 15);
        }
#pragma unroll
        for (int j = 0; j < 4; ++j) {
            const bhalf* p = Bb + (wn + j * 16 + lrow) * LDA + kh;
            v8bf lo = *(const v8bf*)p;
            v8bf hi = *(const v8bf*)(p + 16);
            bfr[j] = __builtin_shufflevector(lo, hi, 0, 1, 2, 3, 4, 5, 6, 7,
                                                     8, 9, 10, 11, 12, 13, 14, 15);
        }
#pragma unroll
        for (int i = 0; i < 2; ++i)
#pragma unroll
            for (int j = 0; j < 4; ++j)
                acc[i][j] = __builtin_amdgcn_wmma_f32_16x16x32_bf16(
                    false, afr[i], false, bfr[j], (short)0, acc[i][j], false, false);

#if !USE_ASYNC_LDS
        if (ks + 1 < NKSTEPS) stash((ks + 1) & 1);     // other buffer; next barrier publishes it
#endif
    }

    // Epilogue: bias + ReLU, store bf16 x. D layout: lane<16 -> M=v, N=lane;
    // lane>=16 -> M=8+v, N=lane-16.
#pragma unroll
    for (int i = 0; i < 2; ++i)
#pragma unroll
        for (int j = 0; j < 4; ++j) {
            const int nn = n0 + wn + j * 16 + lrow;
            const int mb = m0 + wm + i * 16 + (lane >> 4) * 8;
            const float bv = bias[nn];
#pragma unroll
            for (int v = 0; v < 8; ++v) {
                float val = acc[i][j][v] + bv;
                val = val > 0.f ? val : 0.f;
                Xh[(mb + v) * HID + nn] = (bhalf)val;
            }
        }
}

// ---------------------------------------------------------------------------
// Kernel 4: fused 1x1 heads + gather. One wave per gathered anchor slot.
// Waves 0..2047 -> pos (conf + 4 offsets); waves 2048..4095 -> neg (conf only).
__global__ __launch_bounds__(256) void k_heads(const bhalf* __restrict__ Xh,
                                               const float* __restrict__ Wc,
                                               const float* __restrict__ bc,
                                               const float* __restrict__ Wr,
                                               const float* __restrict__ br,
                                               const int* __restrict__ pos,
                                               const int* __restrict__ neg,
                                               float* __restrict__ out) {
    const int gw   = blockIdx.x * 8 + (threadIdx.x >> 5);
    const int lane = threadIdx.x & 31;
    const bool isPos = gw < 2048;
    const int i   = isPos ? gw : (gw - 2048);
    const int idx = isPos ? pos[i] : neg[i];
    const int p = idx / 9;
    const int a = idx - p * 9;

    const bhalf* xr = Xh + p * HID;
    const float* wc = Wc + a * HID;
    float s0 = 0.f, s1 = 0.f, s2 = 0.f, s3 = 0.f, s4 = 0.f;

    if (isPos) {
        const float* w0 = Wr + (a * 4 + 0) * HID;
        const float* w1 = Wr + (a * 4 + 1) * HID;
        const float* w2 = Wr + (a * 4 + 2) * HID;
        const float* w3 = Wr + (a * 4 + 3) * HID;
#pragma unroll 4
        for (int j = 0; j < 16; ++j) {
            int c = j * 32 + lane;            // lane-interleaved: coalesced
            float xv = (float)xr[c];
            s0 += xv * wc[c];
            s1 += xv * w0[c];
            s2 += xv * w1[c];
            s3 += xv * w2[c];
            s4 += xv * w3[c];
        }
    } else {
#pragma unroll 4
        for (int j = 0; j < 16; ++j) {
            int c = j * 32 + lane;
            s0 += (float)xr[c] * wc[c];
        }
    }

    for (int o = 16; o > 0; o >>= 1) {        // wave32 tree reduction
        s0 += __shfl_xor(s0, o, 32);
        if (isPos) {
            s1 += __shfl_xor(s1, o, 32);
            s2 += __shfl_xor(s2, o, 32);
            s3 += __shfl_xor(s3, o, 32);
            s4 += __shfl_xor(s4, o, 32);
        }
    }

    if (lane == 0) {
        if (isPos) {
            out[i] = s0 + bc[a];
            out[4096 + i * 4 + 0] = s1 + br[a * 4 + 0];
            out[4096 + i * 4 + 1] = s2 + br[a * 4 + 1];
            out[4096 + i * 4 + 2] = s3 + br[a * 4 + 2];
            out[4096 + i * 4 + 3] = s4 + br[a * 4 + 3];
        } else {
            out[2048 + i] = s0 + bc[a];
        }
    }
}

// ---------------------------------------------------------------------------
extern "C" void kernel_launch(void* const* d_in, const int* in_sizes, int n_in,
                              void* d_out, int out_size, void* d_ws, size_t ws_size,
                              hipStream_t stream) {
    (void)in_sizes; (void)n_in; (void)out_size; (void)ws_size;
    const float* fm  = (const float*)d_in[0];
    const float* Wh  = (const float*)d_in[1];
    const float* bh  = (const float*)d_in[2];
    const float* Wc  = (const float*)d_in[3];
    const float* bc  = (const float*)d_in[4];
    const float* Wr  = (const float*)d_in[5];
    const float* br  = (const float*)d_in[6];
    const int*   pos = (const int*)d_in[7];
    const int*   neg = (const int*)d_in[8];
    float* out = (float*)d_out;

    // Workspace layout (all 16B aligned):
    //   Apad : 8*34*34*2048 bf16 = 37,879,808 B
    //   Wpk  : 9*2048*512  bf16 = 18,874,368 B
    //   Xh   : 8192*512    bf16 =  8,388,608 B   (total ~62 MiB)
    char* ws = (char*)d_ws;
    const size_t APAD_ELEMS = (size_t)BATCH * HP * WP * CIN;       // 18,939,904
    const size_t WPK_ELEMS  = (size_t)9 * CIN * HID;               //  9,437,184
    bhalf* Apad = (bhalf*)ws;
    bhalf* Wpk  = (bhalf*)(ws + APAD_ELEMS * 2);
    bhalf* Xh   = (bhalf*)(ws + APAD_ELEMS * 2 + WPK_ELEMS * 2);

    const int n16 = (int)(APAD_ELEMS * 2 / 16);                    // 2,367,488
    k_zero<<<(n16 + 255) / 256, 256, 0, stream>>>((uint4*)Apad, n16);
    k_pack_input<<<8192, 256, 0, stream>>>(fm, Apad);
    k_pack_weight<<<(int)(WPK_ELEMS / 256), 256, 0, stream>>>(Wh, Wpk);
    k_conv1_wmma<<<dim3(64, 4, 1), 256, 0, stream>>>(Apad, Wpk, bh, Xh);
    k_heads<<<512, 256, 0, stream>>>(Xh, Wc, bc, Wr, br, pos, neg, out);
}
